// Predictor_46136538693970
// MI455X (gfx1250) — compile-verified, matrix operand order checked by reading, not run on hardware
//
#include <hip/hip_runtime.h>
#include <hip/hip_bf16.h>
#include <math.h>
#include <stdint.h>

#define B_   1024
#define DIM_ 64
#define S_   256
#define T_   255
#define H_   128
#define G_   384   // 3*H
#define CS_  500

typedef __attribute__((ext_vector_type(16))) _Float16 v16h;
typedef __attribute__((ext_vector_type(8)))  _Float16 v8h;
typedef __attribute__((ext_vector_type(8)))  float    v8f;
typedef __attribute__((ext_vector_type(4)))  unsigned int v4u;
typedef __attribute__((ext_vector_type(8)))  int v8i;
typedef __attribute__((ext_vector_type(4)))  int v4i;

__device__ __forceinline__ float leaky(float v) { return v > 0.f ? v : 0.01f * v; }

// K index inside a 32-wide K tile for fragment element j on this lane
// (16-bit A/B layout, CDNA5 ISA 7.12.2):
// lanes 0-15 hold K 0..7 (v0..3) and 16..23 (v4..7); lanes 16-31 hold 8..15 / 24..31.
__device__ __forceinline__ int klocal(int lane, int j) {
  return ((lane >> 4) << 3) + (j & 7) + ((j >> 3) << 4);
}

// Load one 16x32 f16 A fragment from LDS (row-major, leading dim ld).
__device__ __forceinline__ v16h load_a_frag(const _Float16* base, int row, int ld,
                                            int k0, int lane) {
  const _Float16* p = base + row * ld + k0 + ((lane >> 4) << 3);
  v8h lo = *(const v8h*)p;
  v8h hi = *(const v8h*)(p + 16);
  v16h a;
#pragma unroll
  for (int j = 0; j < 8; ++j) { a[j] = lo[j]; a[8 + j] = hi[j]; }
  return a;
}

// Load one 32x16 f16 B fragment for output column g from a row-major f32
// weight matrix W[g][k] (ldk = K). B[k][n] = W[n][k].
__device__ __forceinline__ v16h load_w_frag(const float* W, int ldk, int g,
                                            int kt, int lane) {
  v16h b;
#pragma unroll
  for (int j = 0; j < 16; ++j) {
    int k = kt * 32 + klocal(lane, j);
    b[j] = (_Float16)W[g * ldk + k];
  }
  return b;
}

// TDM: async load a d1 x d0 f16 tile (row stride stride0 elements) into LDS.
__device__ __forceinline__ void tdm_load_2d(const _Float16* gptr, unsigned lds_off,
                                            unsigned d0, unsigned d1,
                                            unsigned stride0) {
  uint64_t ga = (uint64_t)(uintptr_t)gptr;
  v4u g0;
  g0[0] = 1u;                                   // count=1, user descriptor
  g0[1] = lds_off;                              // lds_addr
  g0[2] = (unsigned)(ga & 0xFFFFFFFFu);         // global_addr[31:0]
  g0[3] = (unsigned)((ga >> 32) & 0x1FFFFFFu)   // global_addr[56:32]
          | 0x80000000u;                        // type=2 ("image") bits[127:126]
  v8i g1;
  g1[0] = (1 << 16);                            // data_size=1 -> 2 bytes/elem
  g1[1] = (int)((d0 & 0xFFFFu) << 16);          // tensor_dim0[15:0] at bits 63:48
  g1[2] = (int)((d0 >> 16) | ((d1 & 0xFFFFu) << 16));  // dim0 hi | dim1 lo
  g1[3] = (int)((d1 >> 16) | ((d0 & 0xFFFFu) << 16));  // dim1 hi | tile_dim0
  g1[4] = (int)(d1 & 0xFFFFu);                  // tile_dim1 | tile_dim2=0
  g1[5] = (int)stride0;                         // tensor_dim0_stride[31:0]
  g1[6] = 0;
  g1[7] = 0;
  v4i z = {0, 0, 0, 0};
#if __has_include(<hip/amd_detail/amd_gfx1250_TDM.h>)
  v8i z8 = {0, 0, 0, 0, 0, 0, 0, 0};
  __builtin_amdgcn_tensor_load_to_lds(g0, g1, z, z, z8, 0);
#else
  __builtin_amdgcn_tensor_load_to_lds(g0, g1, z, z, 0);
#endif
}

// ---------------------------------------------------------------------------
// K1: gather snippet rows + transpose-concat into xcT (B*T, 128) f16.
// xcT[(b*T+t)*128 + i] = i<64 ? x[b,i,t] : snippet_list[i-64, snippet[b,i-64], t]
// ---------------------------------------------------------------------------
__global__ __launch_bounds__(256) void gather_pack_kernel(
    const float* __restrict__ x, const int* __restrict__ snippet,
    const float* __restrict__ slist, _Float16* __restrict__ xcT) {
  size_t idx = (size_t)blockIdx.x * 256 + threadIdx.x;
  size_t total = (size_t)B_ * T_ * 128;
  if (idx >= total) return;
  int i = (int)(idx & 127);
  size_t bt = idx >> 7;
  int t = (int)(bt % T_);
  int b = (int)(bt / T_);
  float v;
  if (i < 64) {
    v = x[((size_t)b * 64 + i) * T_ + t];
  } else {
    int d = i - 64;
    int sn = snippet[b * 64 + d];
    v = slist[((size_t)d * CS_ + sn) * S_ + t];
  }
  xcT[idx] = (_Float16)v;
}

__global__ __launch_bounds__(256) void last_gather_kernel(
    const int* __restrict__ snippet, const float* __restrict__ slist,
    float* __restrict__ lastv) {
  int idx = blockIdx.x * 256 + threadIdx.x;
  if (idx >= B_ * 64) return;
  int b = idx >> 6, d = idx & 63;
  int sn = snippet[b * 64 + d];
  lastv[idx] = slist[((size_t)d * CS_ + sn) * S_ + (S_ - 1)];
}

// ---------------------------------------------------------------------------
// K2: xg = xcT(BT x 128) @ W_ih^T(128 x 384) + b_ih, stored f16.
// Persistent WGs; W_ih fragments in VGPRs; A tiles arrive via TDM (overlapped
// with the previous block's epilogue); output staged in LDS for coalesced
// b128 stores.
// ---------------------------------------------------------------------------
__global__ __launch_bounds__(256) void input_gemm_kernel(
    const _Float16* __restrict__ xcT, const float* __restrict__ W_ih,
    const float* __restrict__ b_ih, _Float16* __restrict__ xg) {
  __shared__ __align__(16) _Float16 As[64 * 128];   // 16 KB input tile
  __shared__ __align__(16) _Float16 Os[32 * G_];    // 24 KB output stage
  int tid = threadIdx.x;
  int lane = tid & 31;
  int wave = tid >> 5;
  int nt0 = wave * 3;  // this wave's 3 N-tiles (of 24)

  v16h wfrag[3][4];
  float bias[3];
#pragma unroll
  for (int i = 0; i < 3; ++i) {
    int g = (nt0 + i) * 16 + (lane & 15);
    bias[i] = b_ih[g];
#pragma unroll
    for (int kt = 0; kt < 4; ++kt) wfrag[i][kt] = load_w_frag(W_ih, 128, g, kt, lane);
  }

  unsigned ldsA = (unsigned)(uintptr_t)&As[0];
  const int nRB = (B_ * T_) / 64;  // 4080 row blocks of 64

  if (tid == 0)  // prefetch first A tile: contiguous 64*128 f16 (1-D)
    tdm_load_2d(xcT + (size_t)blockIdx.x * 64 * 128, ldsA, 64 * 128, 1, 64 * 128);

  for (int rb = blockIdx.x; rb < nRB; rb += gridDim.x) {
    if (tid == 0) __builtin_amdgcn_s_wait_tensorcnt(0);
    __syncthreads();  // A tile visible to all waves

    v8f cacc[4][3];
#pragma unroll
    for (int mt = 0; mt < 4; ++mt) {
      v16h afrag[4];
#pragma unroll
      for (int kt = 0; kt < 4; ++kt)
        afrag[kt] = load_a_frag(As, mt * 16 + (lane & 15), 128, kt * 32, lane);
#pragma unroll
      for (int i = 0; i < 3; ++i) {
        v8f c = {};
#pragma unroll
        for (int kt = 0; kt < 4; ++kt)
          c = __builtin_amdgcn_wmma_f32_16x16x32_f16(false, afrag[kt], false,
                                                     wfrag[i][kt], (short)0, c,
                                                     false, false);
        cacc[mt][i] = c;
      }
    }
    __syncthreads();  // all waves done reading As

    int nrb = rb + gridDim.x;
    if (tid == 0 && nrb < nRB)  // overlap next A-tile DMA with epilogue
      tdm_load_2d(xcT + (size_t)nrb * 64 * 128, ldsA, 64 * 128, 1, 64 * 128);

    // Epilogue in two 32-row halves: stage f16 in LDS, copy out coalesced.
    int n = nt0 * 16 + (lane & 15);
    int mb = (lane >> 4) << 3;
#pragma unroll
    for (int half = 0; half < 2; ++half) {
#pragma unroll
      for (int h = 0; h < 2; ++h) {
        int mt = half * 2 + h;
#pragma unroll
        for (int i = 0; i < 3; ++i) {
#pragma unroll
          for (int r = 0; r < 8; ++r)
            Os[(h * 16 + mb + r) * G_ + n + i * 16] =
                (_Float16)(cacc[mt][i][r] + bias[i]);
        }
      }
      __syncthreads();
      const uint4* src = (const uint4*)Os;
      uint4* dst = (uint4*)(xg + ((size_t)rb * 64 + half * 32) * G_);
      for (int i = tid; i < (32 * G_) / 8; i += 256) dst[i] = src[i];
      __syncthreads();
    }
  }
}

// ---------------------------------------------------------------------------
// K3: persistent GRU scan (16 batch rows per WG) + tail MLP.
// TDM double-buffers the per-step xg tile; hg via WMMA; gates via VALU.
// ---------------------------------------------------------------------------
__global__ __launch_bounds__(256) void gru_mlp_kernel(
    const _Float16* __restrict__ xg, const float* __restrict__ W_hh,
    const float* __restrict__ b_hh, const float* __restrict__ lastv,
    const float* __restrict__ W1, const float* __restrict__ b1,
    const float* __restrict__ W2, const float* __restrict__ b2,
    const float* __restrict__ W3, const float* __restrict__ b3,
    const float* __restrict__ Wl, const float* __restrict__ bl,
    float* __restrict__ out) {
  __shared__ __align__(16) _Float16 hS[16 * 128];
  __shared__ __align__(16) _Float16 xgS[2][16 * G_];
  __shared__ __align__(16) float hgS[16 * G_];
  __shared__ float bhS[G_];

  int tid = threadIdx.x;
  int lane = tid & 31;
  int wave = tid >> 5;
  int nt0 = wave * 3;
  int b0 = blockIdx.x * 16;

  for (int i = tid; i < 16 * 128; i += 256) hS[i] = (_Float16)0.f;
  for (int i = tid; i < G_; i += 256) bhS[i] = b_hh[i];

  v16h wfrag[3][4];
#pragma unroll
  for (int i = 0; i < 3; ++i) {
    int g = (nt0 + i) * 16 + (lane & 15);
#pragma unroll
    for (int kt = 0; kt < 4; ++kt) wfrag[i][kt] = load_w_frag(W_hh, 128, g, kt, lane);
  }

  unsigned lds0 = (unsigned)(uintptr_t)&xgS[0][0];
  unsigned lds1 = (unsigned)(uintptr_t)&xgS[1][0];
  const _Float16* xg_blk = xg + (size_t)b0 * T_ * G_;

  if (tid == 0) tdm_load_2d(xg_blk, lds0, G_, 16, T_ * G_);  // prefetch t=0
  __syncthreads();

  for (int t = 0; t < T_; ++t) {
    if (tid == 0) {
      if (t + 1 < T_) {
        tdm_load_2d(xg_blk + (size_t)(t + 1) * G_, ((t + 1) & 1) ? lds1 : lds0,
                    G_, 16, T_ * G_);
        __builtin_amdgcn_s_wait_tensorcnt(1);  // xg_t done, t+1 in flight
      } else {
        __builtin_amdgcn_s_wait_tensorcnt(0);
      }
    }
    __syncthreads();  // xg_t visible; h from step t-1 visible
    const _Float16* xgt = (t & 1) ? &xgS[1][0] : &xgS[0][0];

    // hg = h @ W_hh^T
    v16h afrag[4];
#pragma unroll
    for (int kt = 0; kt < 4; ++kt)
      afrag[kt] = load_a_frag(hS, lane & 15, 128, kt * 32, lane);
#pragma unroll
    for (int i = 0; i < 3; ++i) {
      v8f c = {};
#pragma unroll
      for (int kt = 0; kt < 4; ++kt)
        c = __builtin_amdgcn_wmma_f32_16x16x32_f16(false, afrag[kt], false,
                                                   wfrag[i][kt], (short)0, c,
                                                   false, false);
      int n = (nt0 + i) * 16 + (lane & 15);
      int mb = (lane >> 4) << 3;
#pragma unroll
      for (int r = 0; r < 8; ++r) hgS[(mb + r) * G_ + n] = c[r];
    }
    __syncthreads();

    // gates: r,z sigmoid; n tanh; h' = (1-z)*n + z*h
    for (int e = tid; e < 16 * 128; e += 256) {
      int m = e >> 7, j = e & 127;
      float xr = (float)xgt[m * G_ + j];
      float xz = (float)xgt[m * G_ + 128 + j];
      float xn = (float)xgt[m * G_ + 256 + j];
      float hr = hgS[m * G_ + j] + bhS[j];
      float hz = hgS[m * G_ + 128 + j] + bhS[128 + j];
      float hn = hgS[m * G_ + 256 + j] + bhS[256 + j];
      float r = 1.f / (1.f + expf(-(xr + hr)));
      float z = 1.f / (1.f + expf(-(xz + hz)));
      float nn = tanhf(xn + r * hn);
      float hold = (float)hS[e];
      hS[e] = (_Float16)((1.f - z) * nn + z * hold);
    }
    __syncthreads();  // protects xg buffer reuse + h for next step
  }

  // ---- tail MLP (reuse hgS as f32 scratch) ----
  float* a1 = hgS;               // 16*64
  float* a2 = hgS + 16 * 64;     // 16*32
  float* a3 = hgS + 16 * 96;     // 16*16
  for (int e = tid; e < 16 * 64; e += 256) {
    int m = e >> 6, o = e & 63;
    float acc = b1[o];
    for (int k = 0; k < 128; ++k) acc += (float)hS[m * 128 + k] * W1[o * 128 + k];
    a1[e] = leaky(acc);
  }
  __syncthreads();
  for (int e = tid; e < 16 * 32; e += 256) {
    int m = e >> 5, o = e & 31;
    float acc = b2[o];
    for (int k = 0; k < 64; ++k) acc += a1[m * 64 + k] * W2[o * 64 + k];
    a2[e] = leaky(acc);
  }
  __syncthreads();
  for (int e = tid; e < 16 * 16; e += 256) {
    int m = e >> 4, o = e & 15;
    float acc = b3[o];
    for (int k = 0; k < 32; ++k) acc += a2[m * 32 + k] * W3[o * 32 + k];
    a3[e] = leaky(acc);
  }
  __syncthreads();
  for (int e = tid; e < 16 * 64; e += 256) {
    int m = e >> 6, o = e & 63;
    float acc = bl[o];
    for (int k = 0; k < 16; ++k) acc += a3[m * 16 + k] * Wl[o * 80 + k];
    for (int k = 0; k < 64; ++k) acc += lastv[(b0 + m) * 64 + k] * Wl[o * 80 + 16 + k];
    out[(size_t)(b0 + m) * 64 + o] = leaky(acc);
  }
}

// ---------------------------------------------------------------------------
extern "C" void kernel_launch(void* const* d_in, const int* in_sizes, int n_in,
                              void* d_out, int out_size, void* d_ws, size_t ws_size,
                              hipStream_t stream) {
  const float* x      = (const float*)d_in[0];
  const int*   snippet= (const int*)d_in[1];
  const float* slist  = (const float*)d_in[2];
  const float* W_ih   = (const float*)d_in[3];
  const float* W_hh   = (const float*)d_in[4];
  const float* b_ih   = (const float*)d_in[5];
  const float* b_hh   = (const float*)d_in[6];
  const float* W1     = (const float*)d_in[7];
  const float* b1     = (const float*)d_in[8];
  const float* W2     = (const float*)d_in[9];
  const float* b2     = (const float*)d_in[10];
  const float* W3     = (const float*)d_in[11];
  const float* b3     = (const float*)d_in[12];
  const float* Wl     = (const float*)d_in[13];
  const float* bl     = (const float*)d_in[14];

  char* ws = (char*)d_ws;
  _Float16* xcT  = (_Float16*)ws;                                   // BT*128 f16
  _Float16* xg   = (_Float16*)(ws + (size_t)B_ * T_ * 128 * 2);     // BT*384 f16
  float*    lastv= (float*)(ws + (size_t)B_ * T_ * 128 * 2 +
                                  (size_t)B_ * T_ * G_ * 2);        // B*64 f32
  float* out = (float*)d_out;

  size_t total = (size_t)B_ * T_ * 128;
  gather_pack_kernel<<<(unsigned)((total + 255) / 256), 256, 0, stream>>>(
      x, snippet, slist, xcT);
  last_gather_kernel<<<(B_ * 64 + 255) / 256, 256, 0, stream>>>(snippet, slist, lastv);
  input_gemm_kernel<<<512, 256, 0, stream>>>(xcT, W_ih, b_ih, xg);
  gru_mlp_kernel<<<B_ / 16, 256, 0, stream>>>(xg, W_hh, b_hh, lastv, W1, b1, W2,
                                              b2, W3, b3, Wl, bl, out);
}
